// Attention_45165876085374
// MI455X (gfx1250) — matrix operand order repaired and checked by reading, not compiled, not run on hardware
//
#include <hip/hip_runtime.h>
#include <hip/hip_bf16.h>

// ---------------------------------------------------------------------------
// GPT-2 attention block for MI455X (gfx1250, wave32, WMMA f32_16x16x32_f16)
//   out0: a [B,T,F] fp32, out1: k [B,H,D,T] fp32, out2: v [B,H,T,D] fp32
// ---------------------------------------------------------------------------

typedef __attribute__((ext_vector_type(16))) _Float16 v16h;
typedef __attribute__((ext_vector_type(8)))  float    v8f;

union Frag16 {       // one WMMA 16-bit operand fragment: 16 halves / lane
  v16h  v;
  float4 f4[2];      // two 16B chunks (the per-lane layout is 2 disjoint runs)
};
static_assert(sizeof(Frag16) == 32, "frag size");

#define WMMA_F32_F16(A, B, C) \
  __builtin_amdgcn_wmma_f32_16x16x32_f16(false, (A), false, (B), (short)0, (C), false, false)

// CDNA5 async global->LDS copy (ASYNCcnt-tracked).  vdst VGPR carries the LDS
// byte address; generic pointers to __shared__ have the LDS offset in [31:0].
__device__ __forceinline__ void async_copy_b128(unsigned lds_addr,
                                                const _Float16* gptr) {
  asm volatile("global_load_async_to_lds_b128 %0, %1, off"
               :: "v"(lds_addr), "v"(gptr) : "memory");
}
#define WAIT_ASYNCCNT(n) asm volatile("s_wait_asynccnt %0" ::"i"(n) : "memory")

constexpr int Bv = 4, Tv = 1024, Fv = 1024, Hv = 16, Pv = 1024, Dv = 64, Sv = 2048;
constexpr int NX  = Bv * Tv * Fv;      // 4M  x elements
constexpr int NW1 = Fv * 3 * Fv;       // 3M  c_attn_w elements
constexpr int NW2 = Fv * Fv;           // 1M  c_proj_w elements
constexpr int NKV = Bv * Hv * Dv * Pv; // 4M past_key / past_value elements

// ---------------------------------------------------------------------------
// Prep 1: x -> f16 ; W_qkv -> f16 transposed [N=3072][K=1024] ; W_proj -> [1024][1024]
// ---------------------------------------------------------------------------
__global__ __launch_bounds__(256) void prep_xw_kernel(
    const float* __restrict__ x, const float* __restrict__ wqkv,
    const float* __restrict__ wproj,
    _Float16* __restrict__ xh, _Float16* __restrict__ wqkvT,
    _Float16* __restrict__ wprojT) {
  const size_t total = (size_t)NX + NW1 + NW2;
  for (size_t i = (size_t)blockIdx.x * blockDim.x + threadIdx.x; i < total;
       i += (size_t)gridDim.x * blockDim.x) {
    if (i < NX) {
      xh[i] = (_Float16)x[i];
    } else if (i < (size_t)NX + NW1) {
      size_t j = i - NX;                 // j = n*1024 + k
      int n = (int)(j >> 10), k = (int)(j & 1023);
      wqkvT[j] = (_Float16)wqkv[(size_t)k * 3072 + n];
    } else {
      size_t j = i - NX - NW1;
      int n = (int)(j >> 10), k = (int)(j & 1023);
      wprojT[j] = (_Float16)wproj[(size_t)k * 1024 + n];
    }
  }
}

// ---------------------------------------------------------------------------
// Prep 2: past_key [B,H,D,P] -> kS f16 [B,H,S,D] (s in [0,P))
//         past_value [B,H,P,D] -> vT f16 [B,H,D,S] (s in [0,P))
// ---------------------------------------------------------------------------
__global__ __launch_bounds__(256) void prep_kv_kernel(
    const float* __restrict__ pk, const float* __restrict__ pv,
    _Float16* __restrict__ kS, _Float16* __restrict__ vT) {
  const size_t total = (size_t)NKV * 2;
  for (size_t i = (size_t)blockIdx.x * blockDim.x + threadIdx.x; i < total;
       i += (size_t)gridDim.x * blockDim.x) {
    if (i < NKV) {
      int d = (int)(i & 63), p = (int)((i >> 6) & 1023), bh = (int)(i >> 16);
      kS[((size_t)bh * Sv + p) * Dv + d] =
          (_Float16)pk[((size_t)bh * Dv + d) * Pv + p];
    } else {
      size_t j = i - NKV;
      int d = (int)(j & 63), p = (int)((j >> 6) & 1023), bh = (int)(j >> 16);
      vT[((size_t)bh * Dv + d) * Sv + p] =
          (_Float16)pv[((size_t)bh * Pv + p) * Dv + d];
    }
  }
}

// ---------------------------------------------------------------------------
// Generic f16 GEMM, f32 accumulate:  C[M,N] = A[M,K] @ BT[N,K]^T + bias
// block = 256 (8 waves), block tile 128x128, wave tile 64m x 32n (4x2 wmma tiles)
// mode 0: QKV epilogue (scatter q/k/v); mode 1: plain store (proj output)
// ---------------------------------------------------------------------------
__global__ __launch_bounds__(256) void gemm16_kernel(
    const _Float16* __restrict__ A, const _Float16* __restrict__ BT,
    const float* __restrict__ bias, int M, int N, int K, int mode,
    float* __restrict__ outC,                                     // mode 1
    _Float16* __restrict__ qh, float* __restrict__ outK,          // mode 0
    _Float16* __restrict__ kS, float* __restrict__ outV,
    _Float16* __restrict__ vT) {
  const int lane = threadIdx.x & 31;
  const int wave = threadIdx.x >> 5;
  const int l16 = lane & 15;
  const int half = lane >> 4;
  const int nbx = N >> 7;
  const int bx = blockIdx.x % nbx;
  const int by = blockIdx.x / nbx;
  const int wm = wave & 1, wn = wave >> 1;
  const int m0 = by * 128 + wm * 64;
  const int n0 = bx * 128 + wn * 32;

  const _Float16* aPtr[4];
#pragma unroll
  for (int i = 0; i < 4; i++) aPtr[i] = A + (size_t)(m0 + i * 16 + l16) * K;
  const _Float16* bPtr[2];
#pragma unroll
  for (int j = 0; j < 2; j++) bPtr[j] = BT + (size_t)(n0 + j * 16 + l16) * K;

  v8f acc[4][2];
#pragma unroll
  for (int i = 0; i < 4; i++)
#pragma unroll
    for (int j = 0; j < 2; j++)
#pragma unroll
      for (int e = 0; e < 8; e++) acc[i][j][e] = 0.f;

  // A-frag per-lane layout (16-bit A 16x32): lanes<16 hold K=[0..7],[16..23],
  // lanes>=16 hold K=[8..15],[24..31]; row M = lane%16.
  const int aoff0 = half ? 8 : 0;
  // B-frag (16-bit B 32x16): lanes<16 hold K=[0..7],[16..23], lanes>=16 K=[8..15],[24..31];
  // column N = lane%16.  BT is n-major so each run is contiguous (16B).
  const int boff0 = half ? 8 : 0;

  for (int kk = 0; kk < K; kk += 32) {
    Frag16 fa[4], fb[2];
#pragma unroll
    for (int i = 0; i < 4; i++) {
      fa[i].f4[0] = *reinterpret_cast<const float4*>(aPtr[i] + kk + aoff0);
      fa[i].f4[1] = *reinterpret_cast<const float4*>(aPtr[i] + kk + 16 + aoff0);
    }
#pragma unroll
    for (int j = 0; j < 2; j++) {
      fb[j].f4[0] = *reinterpret_cast<const float4*>(bPtr[j] + kk + boff0);
      fb[j].f4[1] = *reinterpret_cast<const float4*>(bPtr[j] + kk + boff0 + 16);
    }
    if (kk + 32 < K) {  // L2 prefetch of next k-chunk (global_prefetch_b8)
      __builtin_prefetch(aPtr[0] + kk + 32, 0, 1);
      __builtin_prefetch(bPtr[0] + kk + 32, 0, 1);
    }
#pragma unroll
    for (int i = 0; i < 4; i++)
#pragma unroll
      for (int j = 0; j < 2; j++)
        acc[i][j] = WMMA_F32_F16(fa[i].v, fb[j].v, acc[i][j]);
  }

  // epilogue.  C layout: vgpr r, lanes<16 -> (M=r, N=lane), lanes>=16 -> (M=8+r).
#pragma unroll
  for (int i = 0; i < 4; i++) {
#pragma unroll
    for (int j = 0; j < 2; j++) {
      const int gn = n0 + j * 16 + l16;
      const float bvl = bias[gn];
      if (mode == 1) {
#pragma unroll
        for (int r = 0; r < 8; r++) {
          const int gm = m0 + i * 16 + r + half * 8;
          outC[(size_t)gm * N + gn] = acc[i][j][r] + bvl;
        }
      } else {
        const int region = gn >> 10;         // 0:q 1:k 2:v
        const int nn = gn & 1023;
        const int h = nn >> 6, d = nn & 63;
#pragma unroll
        for (int r = 0; r < 8; r++) {
          const int gm = m0 + i * 16 + r + half * 8;
          const int b = gm >> 10, t = gm & 1023;
          const int bh = b * Hv + h;
          const float val = acc[i][j][r] + bvl;
          if (region == 0) {
            // fold 1/sqrt(D)=1/8 into q
            qh[((size_t)bh * Tv + t) * Dv + d] = (_Float16)(val * 0.125f);
          } else if (region == 1) {
            outK[((size_t)bh * Dv + d) * Tv + t] = val;              // [B,H,D,T]
            kS[((size_t)bh * Sv + Pv + t) * Dv + d] = (_Float16)val; // s-major cache
          } else {
            outV[((size_t)bh * Tv + t) * Dv + d] = val;              // [B,H,T,D]
            vT[((size_t)bh * Dv + d) * Sv + Pv + t] = (_Float16)val; // d-major cache
          }
        }
      }
    }
  }
}

// ---------------------------------------------------------------------------
// Flash attention.  block = 128 (4 waves); all 4 waves share one (b,h), each
// owns 16 query rows.  S is processed in 64-column chunks:
//   - K/V chunk tiles (8KB each) are async-copied global->LDS, double-buffered
//     (ASYNCcnt + barrier), shared by all 4 waves.
//   - scores q[16x64] @ K^T -> 4 C tiles; online softmax with per-row stats
//     replicated across 16-lane halves (xor-shuffle reductions).
//   - probs go C-layout -> LDS -> A-layout; P @ V accumulates 4 O tiles.
// ---------------------------------------------------------------------------
__global__ __launch_bounds__(128) void attn_kernel(
    const _Float16* __restrict__ qh, const _Float16* __restrict__ kS,
    const _Float16* __restrict__ vT, _Float16* __restrict__ ah) {
  __shared__ _Float16 ldsK[2][64][64];   // [buf][s-local][d]       8KB x2
  __shared__ _Float16 ldsV[2][64][64];   // [buf][d][s-local]       8KB x2
  __shared__ _Float16 pLds[4][16][64];   // per-wave probs staging  8KB

  const int tid = threadIdx.x;
  const int lane = tid & 31;
  const int wave = tid >> 5;
  const int l16 = lane & 15, half = lane >> 4;
  const int bh = blockIdx.x >> 4;                       // 16 blocks per (b,h)
  const int t0 = ((blockIdx.x & 15) * 4 + wave) * 16;
  const int b = bh >> 4, h = bh & 15;

  const int aoff0 = half ? 8 : 0;
  const int boff0 = half ? 8 : 0;

  const _Float16* kBase = kS + (size_t)bh * Sv * Dv;    // [s][d]
  const _Float16* vBase = vT + (size_t)bh * Dv * Sv;    // [d][s]

  // cooperative async copy of one 64-wide S chunk (K tile + V tile)
  auto issue_chunk = [&](int s0, int buf) {
#pragma unroll
    for (int c = 0; c < 4; c++) {
      const int idx = tid + c * 128;     // 0..511 16B-chunks per tile
      const int row = idx >> 3;          // 0..63
      const int col = (idx & 7) * 8;     // halves
      async_copy_b128((unsigned)(size_t)&ldsK[buf][row][col],
                      kBase + (size_t)(s0 + row) * Dv + col);
      async_copy_b128((unsigned)(size_t)&ldsV[buf][row][col],
                      vBase + (size_t)row * Sv + s0 + col);
    }
  };

  // Q A-fragments for d-chunks [0,32) and [32,64); reused for all S chunks.
  const _Float16* qrow = qh + ((size_t)bh * Tv + t0 + l16) * Dv;
  Frag16 aQ[2];
#pragma unroll
  for (int dc = 0; dc < 2; dc++) {
    aQ[dc].f4[0] = *reinterpret_cast<const float4*>(qrow + dc * 32 + aoff0);
    aQ[dc].f4[1] = *reinterpret_cast<const float4*>(qrow + dc * 32 + 16 + aoff0);
  }

  // Per-row stats: index r maps to row M = r + 8*half (matches C layout);
  // values are replicated across each 16-lane half after xor-reductions.
  float rowmax[8], rowsum[8];
  v8f o[4];
#pragma unroll
  for (int r = 0; r < 8; r++) { rowmax[r] = -3.0e38f; rowsum[r] = 0.f; }
#pragma unroll
  for (int nt = 0; nt < 4; nt++)
#pragma unroll
    for (int e = 0; e < 8; e++) o[nt][e] = 0.f;

  constexpr float L2E = 1.44269504088896341f;
  constexpr int NCHUNK = Sv / 64;        // 32

  issue_chunk(0, 0);
  for (int ci = 0; ci < NCHUNK; ci++) {
    const int buf = ci & 1;
    if (ci + 1 < NCHUNK) {
      issue_chunk((ci + 1) * 64, buf ^ 1);
      WAIT_ASYNCCNT(8);                  // this chunk's 8 copies have landed
    } else {
      WAIT_ASYNCCNT(0);
    }
    __syncthreads();                     // everyone's copies visible

    // ---- scores: q[16x64] @ K^T chunk -> four 16x16 tiles ------------------
    v8f sc[4];
#pragma unroll
    for (int st = 0; st < 4; st++) {
      const _Float16* kp = &ldsK[buf][st * 16 + l16][0];
      Frag16 fb0, fb1;
      fb0.f4[0] = *reinterpret_cast<const float4*>(kp + boff0);
      fb0.f4[1] = *reinterpret_cast<const float4*>(kp + boff0 + 16);
      fb1.f4[0] = *reinterpret_cast<const float4*>(kp + 32 + boff0);
      fb1.f4[1] = *reinterpret_cast<const float4*>(kp + 32 + boff0 + 16);
      v8f z;
#pragma unroll
      for (int e = 0; e < 8; e++) z[e] = 0.f;
      z = WMMA_F32_F16(aQ[0].v, fb0.v, z);
      sc[st] = WMMA_F32_F16(aQ[1].v, fb1.v, z);
    }

    // ---- online softmax ----------------------------------------------------
#pragma unroll
    for (int r = 0; r < 8; r++) {
      float cm = fmaxf(fmaxf(sc[0][r], sc[1][r]), fmaxf(sc[2][r], sc[3][r]));
      cm = fmaxf(cm, __shfl_xor(cm, 1, 32));
      cm = fmaxf(cm, __shfl_xor(cm, 2, 32));
      cm = fmaxf(cm, __shfl_xor(cm, 4, 32));
      cm = fmaxf(cm, __shfl_xor(cm, 8, 32));
      const float nm = fmaxf(rowmax[r], cm);
      const float corr = exp2f((rowmax[r] - nm) * L2E);
      rowmax[r] = nm;
      float p[4], ps = 0.f;
#pragma unroll
      for (int st = 0; st < 4; st++) {
        p[st] = exp2f((sc[st][r] - nm) * L2E);
        ps += p[st];
      }
      ps += __shfl_xor(ps, 1, 32);
      ps += __shfl_xor(ps, 2, 32);
      ps += __shfl_xor(ps, 4, 32);
      ps += __shfl_xor(ps, 8, 32);
      rowsum[r] = rowsum[r] * corr + ps;
#pragma unroll
      for (int nt = 0; nt < 4; nt++) o[nt][r] *= corr;
      const int Mrow = r + half * 8;
#pragma unroll
      for (int st = 0; st < 4; st++)
        pLds[wave][Mrow][st * 16 + l16] = (_Float16)p[st];
    }
    // same-wave DS ordering guarantees our stores land before our reads; the
    // probs slice is private per wave, so no extra block barrier needed here.

    // ---- re-load probs as two A-fragments (s-K chunks [0,32),[32,64)) ------
    Frag16 pA[2];
#pragma unroll
    for (int kc = 0; kc < 2; kc++) {
      const _Float16* pp = &pLds[wave][l16][kc * 32];
      pA[kc].f4[0] = *reinterpret_cast<const float4*>(pp + aoff0);
      pA[kc].f4[1] = *reinterpret_cast<const float4*>(pp + 16 + aoff0);
    }
    // ---- P @ V chunk -------------------------------------------------------
#pragma unroll
    for (int nt = 0; nt < 4; nt++) {
#pragma unroll
      for (int kc = 0; kc < 2; kc++) {
        const _Float16* vp = &ldsV[buf][nt * 16 + l16][kc * 32 + boff0];
        Frag16 fv;
        fv.f4[0] = *reinterpret_cast<const float4*>(vp);
        fv.f4[1] = *reinterpret_cast<const float4*>(vp + 16);
        o[nt] = WMMA_F32_F16(pA[kc].v, fv.v, o[nt]);
      }
    }
    __syncthreads();                     // all waves done with buf before reuse
  }

  // ---- normalize + write merged-head f16 activation ------------------------
#pragma unroll
  for (int nt = 0; nt < 4; nt++)
#pragma unroll
    for (int r = 0; r < 8; r++) {
      const int t = t0 + r + half * 8;
      const int col = h * 64 + nt * 16 + l16;
      ah[((size_t)b * Tv + t) * Fv + col] = (_Float16)(o[nt][r] / rowsum[r]);
    }
}

// ---------------------------------------------------------------------------
extern "C" void kernel_launch(void* const* d_in, const int* in_sizes, int n_in,
                              void* d_out, int out_size, void* d_ws,
                              size_t ws_size, hipStream_t stream) {
  (void)in_sizes; (void)n_in; (void)out_size; (void)ws_size;
  const float* x       = (const float*)d_in[0];
  const float* past_k  = (const float*)d_in[1];
  const float* past_v  = (const float*)d_in[2];
  const float* attn_w  = (const float*)d_in[3];
  const float* attn_b  = (const float*)d_in[4];
  const float* proj_w  = (const float*)d_in[5];
  const float* proj_b  = (const float*)d_in[6];

  float* out_a = (float*)d_out;                       // [B,T,F]
  float* out_k = out_a + (size_t)Bv * Tv * Fv;        // [B,H,D,T]
  float* out_v = out_k + (size_t)Bv * Hv * Dv * Tv;   // [B,H,T,D]

  char* ws = (char*)d_ws;                       // 64 MB total layout:
  _Float16* xh     = (_Float16*)(ws);                         //  0.. 8MB  x f16
  _Float16* wqkvT  = (_Float16*)(ws + (((size_t)8)  << 20));  //  8..14MB  Wqkv^T
  _Float16* wprojT = (_Float16*)(ws + (((size_t)14) << 20));  // 14..16MB  Wproj^T
  _Float16* qh     = (_Float16*)(ws + (((size_t)16) << 20));  // 16..24MB  q/8 f16
  _Float16* kSc    = (_Float16*)(ws + (((size_t)24) << 20));  // 24..40MB  K cache [B,H,S,D]
  _Float16* vTc    = (_Float16*)(ws + (((size_t)40) << 20));  // 40..56MB  V cache [B,H,D,S]
  _Float16* ah     = (_Float16*)(ws + (((size_t)56) << 20));  // 56..64MB  attn f16

  prep_xw_kernel<<<4096, 256, 0, stream>>>(x, attn_w, proj_w, xh, wqkvT, wprojT);
  prep_kv_kernel<<<4096, 256, 0, stream>>>(past_k, past_v, kSc, vTc);

  // QKV GEMM: M=4096, N=3072, K=1024, scatter epilogue
  {
    const int M = Bv * Tv, N = 3 * Fv, K = Fv;
    dim3 grid((M / 128) * (N / 128));
    gemm16_kernel<<<grid, 256, 0, stream>>>(xh, wqkvT, attn_b, M, N, K,
                                            /*mode=*/0, nullptr, qh, out_k,
                                            kSc, out_v, vTc);
  }

  // Flash attention over S=2048
  attn_kernel<<<Bv * Hv * (Tv / 64), 128, 0, stream>>>(qh, kSc, vTc, ah);

  // Projection GEMM: M=4096, N=1024, K=1024, plain fp32 output + bias
  {
    const int M = Bv * Tv, N = Fv, K = Fv;
    dim3 grid((M / 128) * (N / 128));
    gemm16_kernel<<<grid, 256, 0, stream>>>(ah, wprojT, proj_b, M, N, K,
                                            /*mode=*/1, out_a, nullptr, nullptr,
                                            nullptr, nullptr, nullptr);
  }
}
